// TinyAutoRegressive_33414845563639
// MI455X (gfx1250) — matrix operand order repaired and checked by reading, not compiled, not run on hardware
//
#include <hip/hip_runtime.h>

typedef __attribute__((ext_vector_type(16))) _Float16 v16h;
typedef __attribute__((ext_vector_type(8)))  _Float16 v8h;
typedef __attribute__((ext_vector_type(8)))  float    v8f;
typedef __attribute__((ext_vector_type(4)))  float    v4f;

#define HIDN 32
#define ROWS_PER_WAVE 16
#define TTILE 32

__device__ __forceinline__ float fast_sigmoid_pre(float u) {
  // sigmoid(u) where u is already the full pre-activation
  return __builtin_amdgcn_rcpf(1.0f + __builtin_amdgcn_exp2f(u * -1.44269504f));
}
__device__ __forceinline__ float fast_tanh(float u) {
  float e = __builtin_amdgcn_exp2f(u * -2.88539008f);   // e^(-2u)
  return (1.0f - e) * __builtin_amdgcn_rcpf(1.0f + e);
}

// NOTE: min-waves-per-EU must stay 1 — this kernel holds ~200 VGPRs of
// fragments/state by design; any occupancy floor forces scratch spills.
__global__ __launch_bounds__(32, 1) void gru_fused_kernel(
    const float* __restrict__ x, const float* __restrict__ h0,
    const float* __restrict__ W_ih, const float* __restrict__ W_hh,
    const float* __restrict__ b_ih, const float* __restrict__ b_hh,
    const float* __restrict__ W_out, const float* __restrict__ b_out,
    float* __restrict__ out, float* __restrict__ h_last, int T)
{
  // one wave per workgroup: everything below is wave-private
  __shared__ alignas(16) _Float16 hstage[ROWS_PER_WAVE * HIDN];
  __shared__ alignas(16) float    xtile[2][TTILE][ROWS_PER_WAVE];

  const int lane = threadIdx.x & 31;
  const int nlo  = lane & 15;        // column-within-tile index
  const int hi   = lane >> 4;        // half-wave select
  const int b0   = blockIdx.x * ROWS_PER_WAVE;

  // ---- constant B fragments: W_hh^T as six [32k x 16n] tiles (f16) ----
  // B layout (16-bit, 32x16): lane L element i holds k = (i<8 ? i : i+8) + 8*hi, n = L&15.
  v16h whh[6];
  #pragma unroll
  for (int j = 0; j < 6; ++j) {
    const float* src = W_hh + (size_t)(16 * j + nlo) * HIDN + 8 * hi; // row G, k block
    #pragma unroll
    for (int i = 0; i < 16; ++i) whh[j][i] = (_Float16)src[(i < 8) ? i : (i + 8)];
  }
  // W_out replicated across all 16 columns -> every lane gets out[m] in D.
  v16h wout;
  #pragma unroll
  for (int i = 0; i < 16; ++i) wout[i] = (_Float16)W_out[8 * hi + ((i < 8) ? i : (i + 8))];

  // per-lane input-projection constants: G = 16*j + nlo
  float wih[6], bc[6];
  #pragma unroll
  for (int j = 0; j < 6; ++j) {
    int G = 16 * j + nlo;
    wih[j] = W_ih[G];
    bc[j]  = b_ih[G] + b_hh[G];
  }
  v8f bsplat;
  {
    const float bo = b_out[0];
    #pragma unroll
    for (int v = 0; v < 8; ++v) bsplat[v] = bo;
  }

  // ---- h state (f32) in D layout: lane L holds (m = v + 8*hi, k = 16*jj + nlo) ----
  float hs0[8], hs1[8];
  #pragma unroll
  for (int v = 0; v < 8; ++v) {
    hs0[v] = h0[(size_t)(b0 + v + 8 * hi) * HIDN + nlo];
    hs1[v] = h0[(size_t)(b0 + v + 8 * hi) * HIDN + 16 + nlo];
  }
  // initial A fragment (16-bit A 16x32 layout): lane L: M = nlo,
  // halves i<8 -> k = i + 8*hi ; i>=8 -> k = i + 8 + 8*hi
  v16h afrag;
  #pragma unroll
  for (int i = 0; i < 16; ++i) {
    int k = ((i < 8) ? i : (i + 8)) + 8 * hi;
    afrag[i] = (_Float16)h0[(size_t)(b0 + nlo) * HIDN + k];
  }

  // ---- x double-buffer plumbing: tile = 16 batch x 32 timesteps ----
  const int bb  = lane >> 1;
  const int ttb = (lane & 1) * 16;
  const float* xrowbase = x + (size_t)(b0 + bb) * T + ttb;

  v4f pf[4];
  #pragma unroll
  for (int q = 0; q < 4; ++q) pf[q] = *(const v4f*)(xrowbase + 4 * q);       // tile 0
  #pragma unroll
  for (int q = 0; q < 4; ++q)
    #pragma unroll
    for (int e = 0; e < 4; ++e)
      xtile[0][ttb + 4 * q + e][bb] = pf[q][e];                              // commit tile 0
  #pragma unroll
  for (int q = 0; q < 4; ++q) pf[q] = *(const v4f*)(xrowbase + TTILE + 4*q); // prefetch tile 1

  float obuf[4][8];

  for (int tg = 0; tg < T; tg += TTILE) {
    const int buf = (tg >> 5) & 1;
    for (int tt = 0; tt < TTILE; tt += 4) {
      #pragma unroll
      for (int p = 0; p < 4; ++p) {
        const int tl = tt + p;
        // x for this lane's 8 batch rows (m = v + 8*hi)
        v4f xa = *(const v4f*)&xtile[buf][tl][8 * hi];
        v4f xb = *(const v4f*)&xtile[buf][tl][8 * hi + 4];

        // fold input projection + biases of r/z gates into the WMMA C operand
        v8f c0, c1, c2, c3;
        #pragma unroll
        for (int v = 0; v < 8; ++v) {
          float xv = (v < 4) ? xa[v & 3] : xb[v & 3];
          c0[v] = xv * wih[0] + bc[0];
          c1[v] = xv * wih[1] + bc[1];
          c2[v] = xv * wih[2] + bc[2];
          c3[v] = xv * wih[3] + bc[3];
        }

        // r/z pre-activations come straight out of the WMMA (D = h@Whh^T + C)
        v8f d0 = __builtin_amdgcn_wmma_f32_16x16x32_f16(false, afrag, false, whh[0], (short)0, c0, false, false);
        v8f d1 = __builtin_amdgcn_wmma_f32_16x16x32_f16(false, afrag, false, whh[1], (short)0, c1, false, false);
        v8f d2 = __builtin_amdgcn_wmma_f32_16x16x32_f16(false, afrag, false, whh[2], (short)0, c2, false, false);
        v8f d3 = __builtin_amdgcn_wmma_f32_16x16x32_f16(false, afrag, false, whh[3], (short)0, c3, false, false);
        v8f z4 = {}, z5 = {};
        v8f d4 = __builtin_amdgcn_wmma_f32_16x16x32_f16(false, afrag, false, whh[4], (short)0, z4, false, false);
        v8f d5 = __builtin_amdgcn_wmma_f32_16x16x32_f16(false, afrag, false, whh[5], (short)0, z5, false, false);

        // lane-local gate math + state update; stage h_t (f16) for transpose
        #pragma unroll
        for (int v = 0; v < 8; ++v) {
          float xv = (v < 4) ? xa[v & 3] : xb[v & 3];
          { // k = nlo  (tiles 0/2/4 = r/z/n low half)
            float rr = fast_sigmoid_pre(d0[v]);
            float zz = fast_sigmoid_pre(d2[v]);
            float nn = fast_tanh(xv * wih[4] + bc[4] + rr * d4[v]);
            float hn = (1.0f - zz) * nn + zz * hs0[v];
            hs0[v] = hn;
            hstage[(v + 8 * hi) * HIDN + nlo] = (_Float16)hn;
          }
          { // k = 16 + nlo (tiles 1/3/5)
            float rr = fast_sigmoid_pre(d1[v]);
            float zz = fast_sigmoid_pre(d3[v]);
            float nn = fast_tanh(xv * wih[5] + bc[5] + rr * d5[v]);
            float hn = (1.0f - zz) * nn + zz * hs1[v];
            hs1[v] = hn;
            hstage[(v + 8 * hi) * HIDN + 16 + nlo] = (_Float16)hn;
          }
        }
        __builtin_amdgcn_wave_barrier();
        // rebuild A fragment for h_t (D -> A transpose through LDS)
        const _Float16* ar = hstage + nlo * HIDN + 8 * hi;
        v8h alo  = *(const v8h*)ar;         // K = 8*hi .. 8*hi+7
        v8h ahi2 = *(const v8h*)(ar + 16);  // K = 16 + 8*hi .. 23 + 8*hi
        afrag = __builtin_shufflevector(alo, ahi2,
                  0,1,2,3,4,5,6,7,8,9,10,11,12,13,14,15);
        __builtin_amdgcn_wave_barrier();

        // fused output projection: 7th WMMA of the step, b_out folded into C
        v8f od = __builtin_amdgcn_wmma_f32_16x16x32_f16(false, afrag, false, wout, (short)0, bsplat, false, false);
        #pragma unroll
        for (int v = 0; v < 8; ++v) obuf[p][v] = od[v];
      }
      // store 4 timesteps of outputs; lanes 0 and 16 jointly cover all 16 rows
      if (nlo == 0) {
        const int t0 = tg + tt;
        #pragma unroll
        for (int v = 0; v < 8; ++v) {
          v4f o = { obuf[0][v], obuf[1][v], obuf[2][v], obuf[3][v] };
          *(v4f*)(out + (size_t)(b0 + v + 8 * hi) * T + t0) = o;
        }
      }
    }
    // rotate x double-buffer: commit prefetched tile, issue next prefetch
    if (tg + TTILE < T) {
      #pragma unroll
      for (int q = 0; q < 4; ++q)
        #pragma unroll
        for (int e = 0; e < 4; ++e)
          xtile[buf ^ 1][ttb + 4 * q + e][bb] = pf[q][e];
      if (tg + 2 * TTILE < T) {
        #pragma unroll
        for (int q = 0; q < 4; ++q)
          pf[q] = *(const v4f*)(xrowbase + tg + 2 * TTILE + 4 * q);
      }
    }
  }

  // final hidden state h_last[b][k]
  #pragma unroll
  for (int v = 0; v < 8; ++v) {
    h_last[(size_t)(b0 + v + 8 * hi) * HIDN + nlo]      = hs0[v];
    h_last[(size_t)(b0 + v + 8 * hi) * HIDN + 16 + nlo] = hs1[v];
  }
}

extern "C" void kernel_launch(void* const* d_in, const int* in_sizes, int n_in,
                              void* d_out, int out_size, void* d_ws, size_t ws_size,
                              hipStream_t stream) {
  (void)n_in; (void)out_size; (void)d_ws; (void)ws_size;
  const float* x     = (const float*)d_in[0];
  const float* h     = (const float*)d_in[1];
  const float* W_ih  = (const float*)d_in[2];
  const float* W_hh  = (const float*)d_in[3];
  const float* b_ih  = (const float*)d_in[4];
  const float* b_hh  = (const float*)d_in[5];
  const float* W_out = (const float*)d_in[6];
  const float* b_out = (const float*)d_in[7];

  const int B = in_sizes[1] / HIDN;     // h is [1, B, 32]
  const int T = in_sizes[0] / B;        // x is [B, T, 1]

  float* out    = (float*)d_out;
  float* h_last = out + (size_t)B * T;  // outputs concatenated: out, then h_last

  // one 16-row wave per workgroup -> scheduler spreads 256 waves across WGPs
  const int blocks = B / ROWS_PER_WAVE; // 4096/16 = 256
  gru_fused_kernel<<<blocks, 32, 0, stream>>>(x, h, W_ih, W_hh, b_ih, b_hh,
                                              W_out, b_out, out, h_last, T);
}
